// HyperbolicAttention_38070590112429
// MI455X (gfx1250) — matrix operand order repaired and checked by reading, not compiled, not run on hardware
//
#include <hip/hip_runtime.h>
#include <hip/hip_bf16.h>
#include <cstdint>

typedef __bf16 bf16_t;
typedef __attribute__((ext_vector_type(8)))  float  v8f;
typedef __attribute__((ext_vector_type(16))) __bf16 v16bf;
typedef __attribute__((ext_vector_type(8)))  __bf16 v8bf;

#define HA_EPS   1e-8f
#define HA_CLAMP 0.99999f

union ABfrag { v16bf v; v8bf h[2]; };

__device__ inline v8f wmma_bf16(v16bf a, v16bf b, v8f c) {
  return __builtin_amdgcn_wmma_f32_16x16x32_bf16(
      /*neg_a=*/false, a, /*neg_b=*/false, b,
      /*c_mod=*/(short)0, c, /*reuse_a=*/false, /*reuse_b=*/false);
}

// Async global->LDS 16-byte copy (ASYNCcnt-tracked, CDNA5 TDM-adjacent path).
__device__ inline void async_copy_b128(uint32_t lds_addr, const void* gaddr) {
  uint64_t ga = (uint64_t)gaddr;
  asm volatile("global_load_async_to_lds_b128 %0, %1, off"
               :: "v"(lds_addr), "v"(ga) : "memory");
}
__device__ inline uint32_t lds_off(const void* p) {
  return (uint32_t)(uintptr_t)p;   // LDS aperture: addr[31:0] is the LDS offset
}

// ---------------- block-wide (128 thread) sum reduction ----------------
__device__ inline float block_sum128(float v) {
  __shared__ float red[4];
  float s = v;
  for (int o = 16; o >= 1; o >>= 1) s += __shfl_xor(s, o, 32);
  int w = threadIdx.x >> 5;
  __syncthreads();
  if ((threadIdx.x & 31) == 0) red[w] = s;
  __syncthreads();
  return red[0] + red[1] + red[2] + red[3];
}

// ---------------- f32 -> bf16 convert ----------------
__global__ void hyatt_cvt_bf16(const float* __restrict__ src, bf16_t* __restrict__ dst, int n) {
  int i = blockIdx.x * blockDim.x + threadIdx.x;
  if (i < n) dst[i] = (bf16_t)src[i];
}

// ---------------- bias ball: y = clamp(expmap0(b)), plus |y|^2 ----------------
__global__ __launch_bounds__(128) void hyatt_biasball(const float* __restrict__ bvec,
                                                      float* __restrict__ bball,
                                                      float* __restrict__ bn2) {
  int t = threadIdx.x;
  float4 v = *(const float4*)(bvec + t * 4);
  float n2 = block_sum128(v.x * v.x + v.y * v.y + v.z * v.z + v.w * v.w);
  float n  = fmaxf(sqrtf(n2), HA_EPS);
  float th = tanhf(fminf(n, 15.f));
  float ny = th * sqrtf(n2) / n;
  float cs = (ny > HA_CLAMP) ? HA_CLAMP / ny : 1.f;
  float s  = (th / n) * cs;
  float4 y; y.x = v.x * s; y.y = v.y * s; y.z = v.z * s; y.w = v.w * s;
  *(float4*)(bball + t * 4) = y;
  float y2 = block_sum128(y.x * y.x + y.y * y.y + y.z * y.z + y.w * y.w);
  if (t == 0) bn2[0] = y2;
}

// ---------------- u = logmap0(x), stored bf16. one block per row ----------------
__global__ __launch_bounds__(128) void hyatt_logmap(const float* __restrict__ x,
                                                    bf16_t* __restrict__ u) {
  int row = blockIdx.x, t = threadIdx.x;
  float4 v = *(const float4*)(x + (size_t)row * 512 + t * 4);
  float n2 = block_sum128(v.x * v.x + v.y * v.y + v.z * v.z + v.w * v.w);
  float n  = fmaxf(sqrtf(n2), HA_EPS);
  float a  = fminf(n, 0.9999999f);
  float at = 0.5f * __logf((1.f + a) / (1.f - a));   // arctanh
  float s  = at / n;
  bf16_t* up = u + (size_t)row * 512 + t * 4;
  up[0] = (bf16_t)(v.x * s); up[1] = (bf16_t)(v.y * s);
  up[2] = (bf16_t)(v.z * s); up[3] = (bf16_t)(v.w * s);
}

// ---------------- WMMA GEMM: C[M,N] = A[M,K] * W[N,K]^T (+bias) ----------------
// grid = (M/128, N/64), block = 128 (4 waves).
// Wave w -> rows [m0+32w, +32) x 64 cols. Shared W tile (64x32) is staged into LDS
// with async-to-LDS loads, double buffered; A fragments come straight from global.
__global__ __launch_bounds__(128) void hyatt_gemm_wmma(const bf16_t* __restrict__ A,
                                                       const bf16_t* __restrict__ W,
                                                       const float* __restrict__ bias,
                                                       float* __restrict__ Cf,
                                                       bf16_t* __restrict__ Cb,
                                                       int M, int N, int K) {
  __shared__ __align__(32) bf16_t wtile[2][64][32];   // [col][k] bf16, 2 x 4KB

  int t    = threadIdx.x;
  int w    = t >> 5;
  int lane = t & 31;
  int l16  = lane & 15;
  int hk   = lane >> 4;
  int m0   = blockIdx.x * 128 + w * 32;
  int n0   = blockIdx.y * 64;

  auto stage_w = [&](int buf, int k0) {
#pragma unroll
    for (int c = 0; c < 2; ++c) {
      int chunk = t + c * 128;              // 256 x 16B chunks = 64x32 bf16
      int col  = chunk >> 2;
      int koff = (chunk & 3) * 8;
      async_copy_b128(lds_off(&wtile[buf][col][koff]),
                      W + (size_t)(n0 + col) * K + k0 + koff);
    }
  };

  v8f acc[2][4] = {};
  const bf16_t* ap0 = A + (size_t)(m0 + l16) * K;
  const bf16_t* ap1 = A + (size_t)(m0 + 16 + l16) * K;

  int nIter = K >> 5;
  stage_w(0, 0);
  for (int it = 0; it < nIter; ++it) {
    int k0 = it << 5;
    if (it + 1 < nIter) {
      stage_w((it + 1) & 1, k0 + 32);
      asm volatile("s_wait_asynccnt 0x2" ::: "memory");  // previous stage landed
    } else {
      asm volatile("s_wait_asynccnt 0x0" ::: "memory");
    }
    __syncthreads();

    if (k0 + 32 < K) __builtin_prefetch(ap0 + k0 + 32, 0, 1);
    ABfrag a0, a1;
    a0.h[0] = *(const v8bf*)(ap0 + k0 + hk * 16);
    a0.h[1] = *(const v8bf*)(ap0 + k0 + hk * 16 + 8);
    a1.h[0] = *(const v8bf*)(ap1 + k0 + hk * 16);
    a1.h[1] = *(const v8bf*)(ap1 + k0 + hk * 16 + 8);

    int cur = it & 1;
#pragma unroll
    for (int jt = 0; jt < 4; ++jt) {
      v16bf bfrag = *(const v16bf*)&wtile[cur][jt * 16 + l16][hk * 16];
      acc[0][jt] = wmma_bf16(a0.v, bfrag, acc[0][jt]);
      acc[1][jt] = wmma_bf16(a1.v, bfrag, acc[1][jt]);
    }
    __syncthreads();   // protect buffer reuse across waves
  }

#pragma unroll
  for (int rg = 0; rg < 2; ++rg)
#pragma unroll
    for (int jt = 0; jt < 4; ++jt) {
      int col = n0 + jt * 16 + l16;
      float badd = bias ? bias[col] : 0.f;
#pragma unroll
      for (int r = 0; r < 8; ++r) {
        int row = m0 + rg * 16 + r + hk * 8;
        float val = acc[rg][jt][r] + badd;
        if (Cf) Cf[(size_t)row * N + col] = val;
        if (Cb) Cb[(size_t)row * N + col] = (bf16_t)val;
      }
    }
}

// ---------------- expmap0 + mobius_add(bias ball) + clamp; bf16 out + per-head |.|^2 ----
__global__ __launch_bounds__(128) void hyatt_hyp_post(const float* __restrict__ E,
                                                      const float* __restrict__ bball,
                                                      const float* __restrict__ bn2p,
                                                      bf16_t* __restrict__ Hout,
                                                      float* __restrict__ n2out) {
  int row = blockIdx.x, t = threadIdx.x;
  float4 v = *(const float4*)(E + (size_t)row * 512 + t * 4);
  float n2 = block_sum128(v.x * v.x + v.y * v.y + v.z * v.z + v.w * v.w);
  float n  = fmaxf(sqrtf(n2), HA_EPS);
  float th = tanhf(fminf(n, 15.f));
  float ny = th * sqrtf(n2) / n;
  float cb = (ny > HA_CLAMP) ? HA_CLAMP / ny : 1.f;
  float sy = (th / n) * cb;
  float4 y; y.x = v.x * sy; y.y = v.y * sy; y.z = v.z * sy; y.w = v.w * sy;
  float x2 = ny * cb; x2 = x2 * x2;

  float4 bb = *(const float4*)(bball + t * 4);
  float xy = block_sum128(y.x * bb.x + y.y * bb.y + y.z * bb.z + y.w * bb.w);
  float y2b = bn2p[0];
  float Ac = 1.f + 2.f * xy + y2b;
  float Bc = 1.f - x2;
  float inv = 1.f / fmaxf(1.f + 2.f * xy + x2 * y2b, HA_EPS);
  float4 res;
  res.x = (Ac * y.x + Bc * bb.x) * inv;
  res.y = (Ac * y.y + Bc * bb.y) * inv;
  res.z = (Ac * y.z + Bc * bb.z) * inv;
  res.w = (Ac * y.w + Bc * bb.w) * inv;
  float r2 = block_sum128(res.x * res.x + res.y * res.y + res.z * res.z + res.w * res.w);
  float rn = fmaxf(sqrtf(r2), HA_EPS);
  float cs2 = (rn > HA_CLAMP) ? HA_CLAMP / rn : 1.f;
  res.x *= cs2; res.y *= cs2; res.z *= cs2; res.w *= cs2;

  bf16_t* hp = Hout + (size_t)row * 512 + t * 4;
  hp[0] = (bf16_t)res.x; hp[1] = (bf16_t)res.y;
  hp[2] = (bf16_t)res.z; hp[3] = (bf16_t)res.w;

  float hp2 = res.x * res.x + res.y * res.y + res.z * res.z + res.w * res.w;
  for (int o = 8; o >= 1; o >>= 1) hp2 += __shfl_xor(hp2, o, 32);
  if ((t & 15) == 0) n2out[(size_t)row * 8 + (t >> 4)] = hp2;
}

// ---------------- V transpose: Vt[b,h,j,s] = Ve[b,s,h*64+j] (bf16) ----------------
__global__ void hyatt_vtrans(const float* __restrict__ Ve, bf16_t* __restrict__ Vt) {
  int bh = blockIdx.x, b = bh >> 3, h = bh & 7;
  for (int idx = threadIdx.x; idx < 64 * 1024; idx += blockDim.x) {
    int j = idx >> 10, s = idx & 1023;
    float v = Ve[((size_t)(b * 1024 + s)) * 512 + h * 64 + j];
    Vt[((size_t)(bh * 64 + j)) * 1024 + s] = (bf16_t)v;
  }
}

// ---------------- fused flash Poincare attention ----------------
// grid = (B*h, T/64), block = 128 (4 waves), wave -> 16 q rows.
// K tile (32x64) and V tile (64x32) are shared by all 4 waves -> staged via
// async-to-LDS double buffering; per-wave P tile transposed through LDS.
__global__ __launch_bounds__(128) void hyatt_flash_attn(const bf16_t* __restrict__ Qh,
                                                        const bf16_t* __restrict__ Kh,
                                                        const bf16_t* __restrict__ Vt,
                                                        const float* __restrict__ q2p,
                                                        const float* __restrict__ k2p,
                                                        bf16_t* __restrict__ attn_bf) {
  const int T = 1024, D = 512, DK = 64;
  __shared__ __align__(32) bf16_t ktile[2][32][64];   // [s][d]   2 x 4KB
  __shared__ __align__(32) bf16_t vtile[2][64][32];   // [j][s]   2 x 4KB
  __shared__ __align__(32) bf16_t lds_p[4][16][32];   // per-wave P transpose

  int t  = threadIdx.x;
  int bh = blockIdx.x, b = bh >> 3, h = bh & 7;
  int w    = t >> 5;
  int lane = t & 31;
  int l16  = lane & 15;
  int hk   = lane >> 4;
  int t0w  = blockIdx.y * 64 + w * 16;

  auto stage_kv = [&](int buf, int s0) {
#pragma unroll
    for (int c = 0; c < 2; ++c) {            // K tile: 32 rows x 64 d
      int chunk = t + c * 128;
      int srow = chunk >> 3, doff = (chunk & 7) * 8;
      async_copy_b128(lds_off(&ktile[buf][srow][doff]),
                      Kh + ((size_t)(b * T + s0 + srow)) * D + h * DK + doff);
    }
#pragma unroll
    for (int c = 0; c < 2; ++c) {            // V tile: 64 j x 32 s
      int chunk = t + c * 128;
      int j = chunk >> 2, soff = (chunk & 3) * 8;
      async_copy_b128(lds_off(&vtile[buf][j][soff]),
                      Vt + ((size_t)(bh * DK + j)) * T + s0 + soff);
    }
  };

  // Q A-fragments: 16x64 = two 16x32 tiles
  const bf16_t* qp = Qh + ((size_t)(b * T + t0w + l16)) * D + h * DK;
  ABfrag aq0, aq1;
  aq0.h[0] = *(const v8bf*)(qp + hk * 16);
  aq0.h[1] = *(const v8bf*)(qp + hk * 16 + 8);
  aq1.h[0] = *(const v8bf*)(qp + 32 + hk * 16);
  aq1.h[1] = *(const v8bf*)(qp + 32 + hk * 16 + 8);

  float q2v[8];
#pragma unroll
  for (int r = 0; r < 8; ++r)
    q2v[r] = q2p[(size_t)(b * T + t0w + r + hk * 8) * 8 + h];

  v8f oacc[4] = {};
  float rowmax[8], rowsum[8];
#pragma unroll
  for (int r = 0; r < 8; ++r) { rowmax[r] = -3.0e38f; rowsum[r] = 0.f; }

  const int nIter = T / 32;
  stage_kv(0, 0);
  for (int it = 0; it < nIter; ++it) {
    int s0 = it * 32;
    if (it + 1 < nIter) {
      stage_kv((it + 1) & 1, s0 + 32);
      asm volatile("s_wait_asynccnt 0x4" ::: "memory");
    } else {
      asm volatile("s_wait_asynccnt 0x0" ::: "memory");
    }
    __syncthreads();
    int cur = it & 1;

    v8f qk[2];
    float k2t[2];
#pragma unroll
    for (int half = 0; half < 2; ++half) {
      v16bf bk0 = *(const v16bf*)&ktile[cur][half * 16 + l16][hk * 16];
      v16bf bk1 = *(const v16bf*)&ktile[cur][half * 16 + l16][32 + hk * 16];
      v8f c = {};
      c = wmma_bf16(aq0.v, bk0, c);
      c = wmma_bf16(aq1.v, bk1, c);
      qk[half] = c;
      k2t[half] = k2p[(size_t)(b * T + s0 + half * 16 + l16) * 8 + h];
    }

    float p[2][8];
#pragma unroll
    for (int half = 0; half < 2; ++half)
#pragma unroll
      for (int r = 0; r < 8; ++r) {
        float xy = -qk[half][r];
        float q2 = q2v[r], k2 = k2t[half];
        float Ac = 1.f + 2.f * xy + k2;
        float Bc = 1.f - q2;
        float num2 = Ac * Ac * q2 + Bc * Bc * k2 + 2.f * Ac * Bc * xy;
        float den  = fmaxf(1.f + 2.f * xy + q2 * k2, HA_EPS);
        float nn = sqrtf(fmaxf(num2, 0.f)) / den;
        nn = fminf(nn, HA_CLAMP);
        float dist = __logf((1.f + nn) / (1.f - nn));   // 2*atanh(nn)
        p[half][r] = dist * -0.125f;                    // -dist/sqrt(dk)
      }

    // online softmax per row (row lives in one VGPR across a 16-lane half)
#pragma unroll
    for (int r = 0; r < 8; ++r) {
      float mc = fmaxf(p[0][r], p[1][r]);
      for (int o = 8; o >= 1; o >>= 1) mc = fmaxf(mc, __shfl_xor(mc, o, 32));
      float mnew = fmaxf(rowmax[r], mc);
      float corr = __expf(rowmax[r] - mnew);
      rowmax[r] = mnew;
      rowsum[r] *= corr;
#pragma unroll
      for (int jt = 0; jt < 4; ++jt) oacc[jt][r] *= corr;
      float e0 = __expf(p[0][r] - mnew);
      float e1 = __expf(p[1][r] - mnew);
      p[0][r] = e0; p[1][r] = e1;
      float ps = e0 + e1;
      for (int o = 8; o >= 1; o >>= 1) ps += __shfl_xor(ps, o, 32);
      rowsum[r] += ps;
    }

    // P tile (C layout) -> per-wave LDS -> A layout bf16 (wave-private, DS in-order)
#pragma unroll
    for (int half = 0; half < 2; ++half)
#pragma unroll
      for (int r = 0; r < 8; ++r)
        lds_p[w][r + hk * 8][l16 + half * 16] = (bf16_t)p[half][r];
    asm volatile("" ::: "memory");   // keep DS store->load order
    ABfrag apf;
    apf.h[0] = *(const v8bf*)&lds_p[w][l16][hk * 16];
    apf.h[1] = *(const v8bf*)&lds_p[w][l16][hk * 16 + 8];
#pragma unroll
    for (int jt = 0; jt < 4; ++jt) {
      v16bf bvf = *(const v16bf*)&vtile[cur][jt * 16 + l16][hk * 16];
      oacc[jt] = wmma_bf16(apf.v, bvf, oacc[jt]);
    }
    __syncthreads();   // protect K/V buffer reuse across waves
  }

#pragma unroll
  for (int r = 0; r < 8; ++r) {
    float inv = 1.f / rowsum[r];
    int trow = t0w + r + hk * 8;
#pragma unroll
    for (int jt = 0; jt < 4; ++jt)
      attn_bf[((size_t)(b * T + trow)) * D + h * DK + jt * 16 + l16] =
          (bf16_t)(oacc[jt][r] * inv);
  }
}

// ------------------------------- host side -------------------------------
extern "C" void kernel_launch(void* const* d_in, const int* in_sizes, int n_in,
                              void* d_out, int out_size, void* d_ws, size_t ws_size,
                              hipStream_t stream) {
  (void)in_sizes; (void)n_in; (void)out_size; (void)ws_size;
  const int B = 4, T = 1024, D = 512;
  const int M = B * T;

  const float* x  = (const float*)d_in[0];
  const float* Wq = (const float*)d_in[1];
  const float* bq = (const float*)d_in[2];
  const float* Wk = (const float*)d_in[3];
  const float* bk = (const float*)d_in[4];
  const float* Wv = (const float*)d_in[5];
  const float* bv = (const float*)d_in[6];
  const float* Wo = (const float*)d_in[7];
  const float* bo = (const float*)d_in[8];
  float* out = (float*)d_out;

  char* ws = (char*)d_ws;
  size_t off = 0;
  auto alloc = [&](size_t bytes) -> void* {
    void* p = ws + off;
    off += (bytes + 255) & ~(size_t)255;
    return p;
  };
  bf16_t* u_bf   = (bf16_t*)alloc((size_t)M * D * 2);
  bf16_t* Wq_bf  = (bf16_t*)alloc((size_t)D * D * 2);
  bf16_t* Wk_bf  = (bf16_t*)alloc((size_t)D * D * 2);
  bf16_t* Wv_bf  = (bf16_t*)alloc((size_t)D * D * 2);
  bf16_t* Wo_bf  = (bf16_t*)alloc((size_t)D * D * 2);
  float*  Qe     = (float*)alloc((size_t)M * D * 4);
  float*  Ke     = (float*)alloc((size_t)M * D * 4);
  float*  Ve     = (float*)alloc((size_t)M * D * 4);
  bf16_t* Qh_bf  = (bf16_t*)alloc((size_t)M * D * 2);
  bf16_t* Kh_bf  = (bf16_t*)alloc((size_t)M * D * 2);
  float*  q2     = (float*)alloc((size_t)M * 8 * 4);
  float*  k2     = (float*)alloc((size_t)M * 8 * 4);
  bf16_t* Vt     = (bf16_t*)alloc((size_t)M * D * 2);
  bf16_t* attnbf = (bf16_t*)alloc((size_t)M * D * 2);
  float*  bqball = (float*)alloc((D + 1) * 4);
  float*  bkball = (float*)alloc((D + 1) * 4);

  // 1) weights -> bf16; bias balls
  int wn = D * D;
  hyatt_cvt_bf16<<<(wn + 255) / 256, 256, 0, stream>>>(Wq, Wq_bf, wn);
  hyatt_cvt_bf16<<<(wn + 255) / 256, 256, 0, stream>>>(Wk, Wk_bf, wn);
  hyatt_cvt_bf16<<<(wn + 255) / 256, 256, 0, stream>>>(Wv, Wv_bf, wn);
  hyatt_cvt_bf16<<<(wn + 255) / 256, 256, 0, stream>>>(Wo, Wo_bf, wn);
  hyatt_biasball<<<1, 128, 0, stream>>>(bq, bqball, bqball + D);
  hyatt_biasball<<<1, 128, 0, stream>>>(bk, bkball, bkball + D);

  // 2) u = logmap0(x)
  hyatt_logmap<<<M, 128, 0, stream>>>(x, u_bf);

  // 3) Qe/Ke/Ve GEMMs (WMMA, async-LDS double-buffered W tiles)
  dim3 ggrid(M / 128, D / 64);
  hyatt_gemm_wmma<<<ggrid, 128, 0, stream>>>(u_bf, Wq_bf, nullptr, Qe, nullptr, M, D, D);
  hyatt_gemm_wmma<<<ggrid, 128, 0, stream>>>(u_bf, Wk_bf, nullptr, Ke, nullptr, M, D, D);
  hyatt_gemm_wmma<<<ggrid, 128, 0, stream>>>(u_bf, Wv_bf, bv,      Ve, nullptr, M, D, D);

  // 4) hyperbolic post (expmap0 + mobius bias + clamp) -> bf16 + per-head norms
  hyatt_hyp_post<<<M, 128, 0, stream>>>(Qe, bqball, bqball + D, Qh_bf, q2);
  hyatt_hyp_post<<<M, 128, 0, stream>>>(Ke, bkball, bkball + D, Kh_bf, k2);

  // 5) V transpose to [b,h,dk,T] bf16
  hyatt_vtrans<<<32, 256, 0, stream>>>(Ve, Vt);

  // 6) fused flash Poincare attention -> attnbf [B*T, D]
  dim3 agrid(B * 8, T / 64);
  hyatt_flash_attn<<<agrid, 128, 0, stream>>>(Qh_bf, Kh_bf, Vt, q2, k2, attnbf);

  // 7) output projection: out = attn @ Wo^T + bo (f32)
  hyatt_gemm_wmma<<<ggrid, 128, 0, stream>>>(attnbf, Wo_bf, bo, out, nullptr, M, D, D);
}
